// Model_33492154974613
// MI455X (gfx1250) — compile-verified, matrix operand order checked by reading, not run on hardware
//
#include <hip/hip_runtime.h>
#include <hip/hip_bf16.h>

// ---------------------------------------------------------------------------
// BigBird encoder forward for MI455X (gfx1250, wave32, WMMA).
// - All dense projections + blocked attention (QK^T, P@V) on
//   v_wmma_f32_16x16x32_bf16 (weights pre-transposed bf16 [N][K]).
// - GEMM weight tiles streamed with GLOBAL_LOAD_ASYNC_TO_LDS_B128
//   (ASYNCcnt / s_wait_asynccnt), double-buffered in LDS.
// ---------------------------------------------------------------------------

typedef __bf16 bf16_t;
typedef __attribute__((ext_vector_type(16))) __bf16 bf16x16;
typedef __attribute__((ext_vector_type(8)))  float  f32x8;
typedef __attribute__((ext_vector_type(4)))  unsigned int u32x4;
typedef __attribute__((ext_vector_type(2)))  unsigned int u32x2;

#define B_    8
#define S_    1024
#define L_    6
#define H_    8
#define DH_   64
#define D_    384
#define QKV_  512
#define QKV3_ 1536
#define MLP_  2048
#define EMB_  512
#define E_    192
#define VC_   164
#define NB_   (S_ / 64)

__device__ __forceinline__ float gelu_f(float x) {
  const float c = 0.7978845608028654f;
  return 0.5f * x * (1.f + tanhf(c * (x + 0.044715f * x * x * x)));
}

__device__ __forceinline__ bf16x16 frag_ld(const bf16_t* p) {
  union { bf16x16 v; u32x4 q[2]; } u;
  u.q[0] = *(const u32x4*)p;
  u.q[1] = *(const u32x4*)(p + 16);
  return u.v;
}

// LDS offset of a generic pointer to __shared__ (ISA 10.2: LDS aperture uses
// addr[31:0] as the wave-relative LDS address).
__device__ __forceinline__ unsigned lds_off(const void* p) {
  return (unsigned)(unsigned long long)p;
}

// ---------------------------------------------------------------------------
// Weight conversion: f32 [K][N] -> bf16 [N][K]
// ---------------------------------------------------------------------------
__global__ void convT_kernel(const float* __restrict__ W, bf16_t* __restrict__ Wt,
                             int K, int N) {
  long n = (long)K * N;
  for (long i = (long)blockIdx.x * blockDim.x + threadIdx.x; i < n;
       i += (long)gridDim.x * blockDim.x) {
    long kr = i / N, c = i % N;
    Wt[c * K + kr] = (bf16_t)W[i];
  }
}

__global__ void conv_kernel(const float* __restrict__ W, bf16_t* __restrict__ O, long n) {
  for (long i = (long)blockIdx.x * blockDim.x + threadIdx.x; i < n;
       i += (long)gridDim.x * blockDim.x)
    O[i] = (bf16_t)W[i];
}

// Concatenate bq|bk|bv per layer into [L][1536]
__global__ void concat_bias_kernel(const float* __restrict__ bq, const float* __restrict__ bk,
                                   const float* __restrict__ bv, float* __restrict__ o) {
  int i = blockIdx.x * 256 + threadIdx.x;
  if (i < L_ * QKV3_) {
    int l = i / QKV3_, c = i % QKV3_;
    float v = (c < QKV_) ? bq[l * QKV_ + c]
            : (c < 2 * QKV_) ? bk[l * QKV_ + c - QKV_]
                             : bv[l * QKV_ + c - 2 * QKV_];
    o[i] = v;
  }
}

// ---------------------------------------------------------------------------
// Embedding
// ---------------------------------------------------------------------------
__global__ __launch_bounds__(128)
void embed_kernel(const int* __restrict__ tc, const int* __restrict__ tw,
                  const float* __restrict__ ce, const float* __restrict__ we,
                  const float* __restrict__ pos, float* __restrict__ x) {
  const long i = blockIdx.x;
  const int  sidx = (int)(i % S_);
  const int  c = tc[i], w = tw[i];
  for (int j = threadIdx.x; j < D_; j += 128) {
    float v = (j < E_) ? ce[(long)c * E_ + j] : we[(long)w * E_ + (j - E_)];
    x[i * D_ + j] = v + pos[(long)sidx * D_ + j];
  }
}

// ---------------------------------------------------------------------------
// LayerNorm over D_=384
// ---------------------------------------------------------------------------
__global__ __launch_bounds__(128)
void layernorm_kernel(const float* __restrict__ x, const float* __restrict__ s,
                      const float* __restrict__ b, float* __restrict__ y) {
  __shared__ float r1[128];
  __shared__ float r2[128];
  const long tok = blockIdx.x;
  const int  t = threadIdx.x;
  const float* xr = x + tok * D_;
  float v0 = xr[t], v1 = xr[t + 128], v2 = xr[t + 256];
  r1[t] = v0 + v1 + v2;
  r2[t] = v0 * v0 + v1 * v1 + v2 * v2;
  __syncthreads();
  for (int off = 64; off > 0; off >>= 1) {
    if (t < off) { r1[t] += r1[t + off]; r2[t] += r2[t + off]; }
    __syncthreads();
  }
  const float mean = r1[0] * (1.f / D_);
  const float var  = r2[0] * (1.f / D_) - mean * mean;
  const float inv  = rsqrtf(var + 1e-6f);
  float* yr = y + tok * D_;
  yr[t]       = (v0 - mean) * inv * s[t]       + b[t];
  yr[t + 128] = (v1 - mean) * inv * s[t + 128] + b[t + 128];
  yr[t + 256] = (v2 - mean) * inv * s[t + 256] + b[t + 256];
}

// ---------------------------------------------------------------------------
// WMMA GEMM:  C[M][ldc] = act( A[M][lda](f32) @ Wt^T + bias ) (+ Res)
// 64x64 tile / 128 threads. A: vectorized clamped f32 loads -> bf16 LDS.
// W: async DMA global->LDS (b128), double-buffered, ASYNCcnt-pipelined.
// ---------------------------------------------------------------------------
template <int ACT, int RES>
__global__ __launch_bounds__(128)
void gemm_wmma_kernel(const float* __restrict__ A, long lda,
                      const bf16_t* __restrict__ Wt,
                      const float* __restrict__ bias,
                      const float* __restrict__ Res, long ldr,
                      float* __restrict__ C, long ldc,
                      int M, int N, int K) {
  __shared__ __align__(16) bf16_t As[64][72];
  __shared__ __align__(16) bf16_t Bs[2][64][72];

  const int tid   = threadIdx.x;
  const int wave  = tid >> 5;
  const int lane  = tid & 31;
  const int lmod  = lane & 15;
  const int lhalf = lane >> 4;
  const int wr    = (wave >> 1) << 5;
  const int wc    = (wave & 1)  << 5;
  const int gRow0 = blockIdx.y * 64;
  const int gCol0 = blockIdx.x * 64;

  float4 aReg[8];

  auto loadA = [&](int k0) {
#pragma unroll
    for (int i = 0; i < 8; ++i) {
      const int f = i * 128 + tid, r = f >> 4, c4 = (f & 15) << 2;
      const int grow = gRow0 + r;
      const int rc = grow < M ? grow : M - 1;
      float4 a = *(const float4*)(A + (long)rc * lda + k0 + c4);
      if (grow >= M) { a.x = 0.f; a.y = 0.f; a.z = 0.f; a.w = 0.f; }
      aReg[i] = a;
    }
  };
  auto storeA = [&]() {
#pragma unroll
    for (int i = 0; i < 8; ++i) {
      const int f = i * 128 + tid, r = f >> 4, c4 = (f & 15) << 2;
      union { bf16_t h[4]; u32x2 d; } pk;
      pk.h[0] = (bf16_t)aReg[i].x; pk.h[1] = (bf16_t)aReg[i].y;
      pk.h[2] = (bf16_t)aReg[i].z; pk.h[3] = (bf16_t)aReg[i].w;
      *(u32x2*)&As[r][c4] = pk.d;
    }
  };
  // Async DMA one 64x64 bf16 weight tile into Bs[buf]: 4 x b128 per thread.
  auto asyncW = [&](int k0, int buf) {
#pragma unroll
    for (int i = 0; i < 4; ++i) {
      const int f = i * 128 + tid, r = f >> 3, c8 = (f & 7) << 3;
      const int gcol = gCol0 + r;
      const int rc = gcol < N ? gcol : N - 1;   // clamp; OOB cols never stored
      const bf16_t* gp = Wt + (long)rc * K + k0 + c8;
      const unsigned lo = lds_off(&Bs[buf][r][c8]);
      asm volatile("global_load_async_to_lds_b128 %0, %1, off"
                   :: "v"(lo), "v"(gp) : "memory");
    }
  };

  f32x8 acc[2][2] = {};
  asyncW(0, 0);
  loadA(0);
  int buf = 0;

  for (int k0 = 0; k0 < K; k0 += 64) {
    storeA();
    const bool more = (k0 + 64) < K;
    if (more) {
      asyncW(k0 + 64, buf ^ 1);                       // prefetch next W tile
      asm volatile("s_wait_asynccnt 0x4" ::: "memory");  // drain current tile
    } else {
      asm volatile("s_wait_asynccnt 0x0" ::: "memory");
    }
    __syncthreads();
    if (more) {
      loadA(k0 + 64);                                  // overlap with compute
      if (k0 + 128 < K) {
        const int ar = gRow0 + (tid >> 4);
        __builtin_prefetch(A + (long)(ar < M ? ar : M - 1) * lda + (k0 + 128) +
                               ((tid & 15) << 2), 0, 1);
      }
    }
#pragma unroll
    for (int kk = 0; kk < 64; kk += 32) {
      const int kb = kk + lhalf * 8;
      bf16x16 afr[2], bfr[2];
      afr[0] = frag_ld(&As[wr + lmod][kb]);
      afr[1] = frag_ld(&As[wr + 16 + lmod][kb]);
      bfr[0] = frag_ld(&Bs[buf][wc + lmod][kb]);
      bfr[1] = frag_ld(&Bs[buf][wc + 16 + lmod][kb]);
#pragma unroll
      for (int mi = 0; mi < 2; ++mi)
#pragma unroll
        for (int ni = 0; ni < 2; ++ni)
          acc[mi][ni] = __builtin_amdgcn_wmma_f32_16x16x32_bf16(
              false, afr[mi], false, bfr[ni], (short)0, acc[mi][ni],
              false, false);
    }
    __syncthreads();
    buf ^= 1;
  }

#pragma unroll
  for (int mi = 0; mi < 2; ++mi) {
#pragma unroll
    for (int ni = 0; ni < 2; ++ni) {
      const int col = gCol0 + wc + 16 * ni + lmod;
#pragma unroll
      for (int r = 0; r < 8; ++r) {
        const int row = gRow0 + wr + 16 * mi + r + 8 * lhalf;
        if (row < M && col < N) {
          float v = acc[mi][ni][r];
          if (bias) v += bias[col];
          if (RES)  v += Res[(long)row * ldr + col];
          if (ACT == 1) v = gelu_f(v);
          C[(long)row * ldc + col] = v;
        }
      }
    }
  }
}

// ---------------------------------------------------------------------------
// Blocked flash attention with WMMA for QK^T and P@V.
// One (query-block, head, batch) per 128-thread workgroup (4 waves).
// qkv: [B*S][1536] with q|k|v interleaved per token.
// ---------------------------------------------------------------------------
__global__ __launch_bounds__(128)
void attention_wmma_kernel(const float* __restrict__ qkv,
                           const unsigned char* __restrict__ bmask,
                           const int* __restrict__ tc,
                           float* __restrict__ out) {
  __shared__ __align__(16) bf16_t Qs[64][72];
  __shared__ __align__(16) bf16_t Ks[64][72];
  __shared__ __align__(16) bf16_t Vt[64][72];   // [d][key]
  __shared__ __align__(16) bf16_t Ps[64][72];
  __shared__ float Ss[64][68];
  __shared__ float mrow[64], lrow[64], crow[64];
  __shared__ int   kval[64];

  const int qb = blockIdx.x, hh = blockIdx.y, bb = blockIdx.z;
  const int tid = threadIdx.x, wave = tid >> 5, lane = tid & 31;
  const int lmod = lane & 15, lhalf = lane >> 4;
  const int wr = (wave >> 1) << 5, wc = (wave & 1) << 5;
  const long tok0 = (long)bb * S_;

#pragma unroll
  for (int i = 0; i < 8; ++i) {
    const int f = i * 128 + tid, r = f >> 4, c4 = (f & 15) << 2;
    const float4 a = *(const float4*)(qkv + (tok0 + qb * 64 + r) * QKV3_ + hh * DH_ + c4);
    union { bf16_t h[4]; u32x2 d; } pk;
    pk.h[0] = (bf16_t)(a.x * 0.125f); pk.h[1] = (bf16_t)(a.y * 0.125f);
    pk.h[2] = (bf16_t)(a.z * 0.125f); pk.h[3] = (bf16_t)(a.w * 0.125f);
    *(u32x2*)&Qs[r][c4] = pk.d;
  }
  if (tid < 64) { mrow[tid] = -3e38f; lrow[tid] = 0.f; }
  f32x8 oacc[2][2] = {};
  __syncthreads();

  for (int kb = 0; kb < NB_; ++kb) {
    if (!bmask[(long)(qb * 64) * S_ + kb * 64]) continue;  // uniform
    __syncthreads();

#pragma unroll
    for (int i = 0; i < 8; ++i) {
      const int f = i * 128 + tid, r = f >> 4, c4 = (f & 15) << 2;
      const long base = (tok0 + kb * 64 + r) * QKV3_ + hh * DH_ + c4;
      const float4 kk4 = *(const float4*)(qkv + base + QKV_);
      union { bf16_t h[4]; u32x2 d; } pk;
      pk.h[0] = (bf16_t)kk4.x; pk.h[1] = (bf16_t)kk4.y;
      pk.h[2] = (bf16_t)kk4.z; pk.h[3] = (bf16_t)kk4.w;
      *(u32x2*)&Ks[r][c4] = pk.d;
      const float4 vv4 = *(const float4*)(qkv + base + 2 * QKV_);
      Vt[c4 + 0][r] = (bf16_t)vv4.x; Vt[c4 + 1][r] = (bf16_t)vv4.y;
      Vt[c4 + 2][r] = (bf16_t)vv4.z; Vt[c4 + 3][r] = (bf16_t)vv4.w;
    }
    if (tid < 64) kval[tid] = tc[tok0 + kb * 64 + tid] > 0;
    __syncthreads();

    f32x8 sacc[2][2] = {};
#pragma unroll
    for (int kk = 0; kk < 64; kk += 32) {
      const int kbse = kk + lhalf * 8;
      bf16x16 afr[2], bfr[2];
      afr[0] = frag_ld(&Qs[wr + lmod][kbse]);
      afr[1] = frag_ld(&Qs[wr + 16 + lmod][kbse]);
      bfr[0] = frag_ld(&Ks[wc + lmod][kbse]);
      bfr[1] = frag_ld(&Ks[wc + 16 + lmod][kbse]);
#pragma unroll
      for (int mi = 0; mi < 2; ++mi)
#pragma unroll
        for (int ni = 0; ni < 2; ++ni)
          sacc[mi][ni] = __builtin_amdgcn_wmma_f32_16x16x32_bf16(
              false, afr[mi], false, bfr[ni], (short)0, sacc[mi][ni],
              false, false);
    }
#pragma unroll
    for (int mi = 0; mi < 2; ++mi)
#pragma unroll
      for (int ni = 0; ni < 2; ++ni)
#pragma unroll
        for (int r = 0; r < 8; ++r)
          Ss[wr + 16 * mi + r + 8 * lhalf][wc + 16 * ni + lmod] = sacc[mi][ni][r];
    __syncthreads();

    if (tid < 64) {
      const float m0 = mrow[tid];
      float bm = m0;
      for (int j = 0; j < 64; ++j) {
        float s = kval[j] ? Ss[tid][j] : -1e9f;
        Ss[tid][j] = s;
        bm = fmaxf(bm, s);
      }
      const float corr = __expf(m0 - bm);
      float ls = lrow[tid] * corr;
      for (int j = 0; j < 64; ++j) {
        const float p = __expf(Ss[tid][j] - bm);
        ls += p;
        Ps[tid][j] = (bf16_t)p;
      }
      mrow[tid] = bm; lrow[tid] = ls; crow[tid] = corr;
    }
    __syncthreads();

#pragma unroll
    for (int mi = 0; mi < 2; ++mi)
#pragma unroll
      for (int ni = 0; ni < 2; ++ni)
#pragma unroll
        for (int r = 0; r < 8; ++r)
          oacc[mi][ni][r] *= crow[wr + 16 * mi + r + 8 * lhalf];
#pragma unroll
    for (int kk = 0; kk < 64; kk += 32) {
      const int kbse = kk + lhalf * 8;
      bf16x16 afr[2], bfr[2];
      afr[0] = frag_ld(&Ps[wr + lmod][kbse]);
      afr[1] = frag_ld(&Ps[wr + 16 + lmod][kbse]);
      bfr[0] = frag_ld(&Vt[wc + lmod][kbse]);
      bfr[1] = frag_ld(&Vt[wc + 16 + lmod][kbse]);
#pragma unroll
      for (int mi = 0; mi < 2; ++mi)
#pragma unroll
        for (int ni = 0; ni < 2; ++ni)
          oacc[mi][ni] = __builtin_amdgcn_wmma_f32_16x16x32_bf16(
              false, afr[mi], false, bfr[ni], (short)0, oacc[mi][ni],
              false, false);
    }
  }

  __syncthreads();
#pragma unroll
  for (int mi = 0; mi < 2; ++mi) {
#pragma unroll
    for (int ni = 0; ni < 2; ++ni) {
      const int col = wc + 16 * ni + lmod;
#pragma unroll
      for (int r = 0; r < 8; ++r) {
        const int row = wr + 16 * mi + r + 8 * lhalf;
        out[(tok0 + qb * 64 + row) * QKV_ + hh * DH_ + col] =
            oacc[mi][ni][r] / lrow[row];
      }
    }
  }
}

// ---------------------------------------------------------------------------
// Host orchestration
// ---------------------------------------------------------------------------
extern "C" void kernel_launch(void* const* d_in, const int* in_sizes, int n_in,
                              void* d_out, int out_size, void* d_ws, size_t ws_size,
                              hipStream_t stream) {
  const int*           tc    = (const int*)d_in[0];
  const int*           tw    = (const int*)d_in[1];
  const unsigned char* bmask = (const unsigned char*)d_in[2];
  const float* ce   = (const float*)d_in[3];
  const float* we   = (const float*)d_in[4];
  const float* pos  = (const float*)d_in[5];
  const float* ln1s = (const float*)d_in[6];
  const float* ln1b = (const float*)d_in[7];
  const float* wq   = (const float*)d_in[8];
  const float* bq   = (const float*)d_in[9];
  const float* wk   = (const float*)d_in[10];
  const float* bk   = (const float*)d_in[11];
  const float* wv   = (const float*)d_in[12];
  const float* bv   = (const float*)d_in[13];
  const float* wo   = (const float*)d_in[14];
  const float* bo   = (const float*)d_in[15];
  const float* ln2s = (const float*)d_in[16];
  const float* ln2b = (const float*)d_in[17];
  const float* w1   = (const float*)d_in[18];
  const float* b1   = (const float*)d_in[19];
  const float* w2   = (const float*)d_in[20];
  const float* b2   = (const float*)d_in[21];
  const float* flns = (const float*)d_in[22];
  const float* flnb = (const float*)d_in[23];
  const float* mw1  = (const float*)d_in[24];
  const float* mb1  = (const float*)d_in[25];
  const float* mw2  = (const float*)d_in[26];
  const float* mb2  = (const float*)d_in[27];
  const float* nw1  = (const float*)d_in[28];
  const float* nb1  = (const float*)d_in[29];
  const float* nw2  = (const float*)d_in[30];
  const float* nb2  = (const float*)d_in[31];
  const float* dw1  = (const float*)d_in[32];
  const float* db1  = (const float*)d_in[33];
  const float* dw2  = (const float*)d_in[34];
  const float* db2  = (const float*)d_in[35];
  const float* rw1  = (const float*)d_in[36];
  const float* rb1  = (const float*)d_in[37];
  const float* rw2  = (const float*)d_in[38];
  const float* rb2  = (const float*)d_in[39];

  char* wp = (char*)d_ws;
  auto alloc = [&](size_t bytes) -> void* {
    void* r = (void*)wp;
    wp += (bytes + 255) & ~(size_t)255;
    return r;
  };

  bf16_t* wqkv_t = (bf16_t*)alloc((size_t)L_ * QKV3_ * D_ * 2);
  bf16_t* wo_t   = (bf16_t*)alloc((size_t)L_ * D_ * QKV_ * 2);
  bf16_t* w1_t   = (bf16_t*)alloc((size_t)L_ * MLP_ * D_ * 2);
  bf16_t* w2_t   = (bf16_t*)alloc((size_t)L_ * D_ * MLP_ * 2);
  bf16_t* mw1t   = (bf16_t*)alloc((size_t)EMB_ * D_ * 2);
  bf16_t* mw2t   = (bf16_t*)alloc((size_t)E_ * EMB_ * 2);
  bf16_t* nw1t   = (bf16_t*)alloc((size_t)EMB_ * D_ * 2);
  bf16_t* nw2t   = (bf16_t*)alloc((size_t)2 * EMB_ * 2);
  bf16_t* dw1t   = (bf16_t*)alloc((size_t)EMB_ * D_ * 2);
  bf16_t* dw2t   = (bf16_t*)alloc((size_t)160 * EMB_ * 2);
  bf16_t* rw1t   = (bf16_t*)alloc((size_t)EMB_ * D_ * 2);
  bf16_t* rw2t   = (bf16_t*)alloc((size_t)85 * EMB_ * 2);
  bf16_t* ce_b   = (bf16_t*)alloc((size_t)VC_ * E_ * 2);
  float*  bqkv   = (float*)alloc((size_t)L_ * QKV3_ * 4);

  const size_t TOK = (size_t)B_ * S_;
  float* x0   = (float*)alloc(TOK * D_ * 4);
  float* x1   = (float*)alloc(TOK * D_ * 4);
  float* hbuf = (float*)alloc(TOK * D_ * 4);
  float* qkvb = (float*)alloc(TOK * QKV3_ * 4);
  float* ao   = (float*)alloc(TOK * QKV_ * 4);
  float* mid  = (float*)alloc(TOK * MLP_ * 4);
  float* hh   = (float*)alloc(TOK * EMB_ * 4);
  float* mpre = (float*)alloc(TOK * E_ * 4);

  const dim3 cb(256);
  const int  cg = 1024;

  for (int l = 0; l < L_; ++l) {
    bf16_t* dst = wqkv_t + (long)l * QKV3_ * D_;
    convT_kernel<<<cg, cb, 0, stream>>>(wq + (long)l * D_ * QKV_, dst,                       D_, QKV_);
    convT_kernel<<<cg, cb, 0, stream>>>(wk + (long)l * D_ * QKV_, dst + (long)QKV_ * D_,     D_, QKV_);
    convT_kernel<<<cg, cb, 0, stream>>>(wv + (long)l * D_ * QKV_, dst + (long)2 * QKV_ * D_, D_, QKV_);
    convT_kernel<<<cg, cb, 0, stream>>>(wo + (long)l * QKV_ * D_, wo_t + (long)l * D_ * QKV_, QKV_, D_);
    convT_kernel<<<cg, cb, 0, stream>>>(w1 + (long)l * D_ * MLP_, w1_t + (long)l * MLP_ * D_, D_, MLP_);
    convT_kernel<<<cg, cb, 0, stream>>>(w2 + (long)l * MLP_ * D_, w2_t + (long)l * D_ * MLP_, MLP_, D_);
  }
  concat_bias_kernel<<<(L_ * QKV3_ + 255) / 256, cb, 0, stream>>>(bq, bk, bv, bqkv);
  convT_kernel<<<cg, cb, 0, stream>>>(mw1, mw1t, D_, EMB_);
  convT_kernel<<<cg, cb, 0, stream>>>(mw2, mw2t, EMB_, E_);
  convT_kernel<<<cg, cb, 0, stream>>>(nw1, nw1t, D_, EMB_);
  convT_kernel<<<cg, cb, 0, stream>>>(nw2, nw2t, EMB_, 2);
  convT_kernel<<<cg, cb, 0, stream>>>(dw1, dw1t, D_, EMB_);
  convT_kernel<<<cg, cb, 0, stream>>>(dw2, dw2t, EMB_, 160);
  convT_kernel<<<cg, cb, 0, stream>>>(rw1, rw1t, D_, EMB_);
  convT_kernel<<<cg, cb, 0, stream>>>(rw2, rw2t, EMB_, 85);
  conv_kernel<<<cg, cb, 0, stream>>>(ce, ce_b, (long)VC_ * E_);

  auto gemm = [&](const float* A, long lda, const bf16_t* Wt, const float* bias,
                  const float* Res, long ldr, float* C, long ldc,
                  int M, int N, int K, int act) {
    dim3 grid((N + 63) / 64, (M + 63) / 64), blk(128);
    if (Res)       gemm_wmma_kernel<0, 1><<<grid, blk, 0, stream>>>(A, lda, Wt, bias, Res, ldr, C, ldc, M, N, K);
    else if (act)  gemm_wmma_kernel<1, 0><<<grid, blk, 0, stream>>>(A, lda, Wt, bias, Res, ldr, C, ldc, M, N, K);
    else           gemm_wmma_kernel<0, 0><<<grid, blk, 0, stream>>>(A, lda, Wt, bias, Res, ldr, C, ldc, M, N, K);
  };

  embed_kernel<<<(int)TOK, 128, 0, stream>>>(tc, tw, ce, we, pos, x0);

  float* xa = x0;
  float* xb = x1;
  const int M = (int)TOK;
  for (int l = 0; l < L_; ++l) {
    layernorm_kernel<<<(int)TOK, 128, 0, stream>>>(xa, ln1s + l * D_, ln1b + l * D_, hbuf);
    gemm(hbuf, D_, wqkv_t + (long)l * QKV3_ * D_, bqkv + l * QKV3_, nullptr, 0,
         qkvb, QKV3_, M, QKV3_, D_, 0);

    attention_wmma_kernel<<<dim3(NB_, H_, B_), 128, 0, stream>>>(
        qkvb, bmask + (long)l * S_ * S_, tc, ao);

    gemm(ao, QKV_, wo_t + (long)l * D_ * QKV_, bo + l * D_, xa, D_, xb, D_, M, D_, QKV_, 0);

    layernorm_kernel<<<(int)TOK, 128, 0, stream>>>(xb, ln2s + l * D_, ln2b + l * D_, hbuf);
    gemm(hbuf, D_, w1_t + (long)l * MLP_ * D_, b1 + l * MLP_, nullptr, 0, mid, MLP_, M, MLP_, D_, 1);
    gemm(mid, MLP_, w2_t + (long)l * D_ * MLP_, b2 + l * D_, xb, D_, xa, D_, M, D_, MLP_, 0);
  }

  layernorm_kernel<<<(int)TOK, 128, 0, stream>>>(xa, flns, flnb, hbuf);

  float* out = (float*)d_out;
  float* out_date = out;
  float* out_reg  = out + (long)B_ * 160;
  float* out_mask = out_reg + (long)B_ * 85;
  float* out_nsp  = out_mask + (long)TOK * VC_;

  gemm(hbuf, D_, mw1t, mb1, nullptr, 0, hh, EMB_, M, EMB_, D_, 1);
  gemm(hh, EMB_, mw2t, mb2, nullptr, 0, mpre, E_, M, E_, EMB_, 0);
  gemm(mpre, E_, ce_b, nullptr, nullptr, 0, out_mask, VC_, M, VC_, E_, 0);

  gemm(hbuf, D_, nw1t, nb1, nullptr, 0, hh, EMB_, M, EMB_, D_, 1);
  gemm(hh, EMB_, nw2t, nb2, nullptr, 0, out_nsp, 2, M, 2, EMB_, 0);

  gemm(hbuf, (long)S_ * D_, dw1t, db1, nullptr, 0, mid, EMB_, B_, EMB_, D_, 1);
  gemm(mid, EMB_, dw2t, db2, nullptr, 0, out_date, 160, B_, 160, EMB_, 0);

  gemm(hbuf, (long)S_ * D_, rw1t, rb1, nullptr, 0, mid, EMB_, B_, EMB_, D_, 1);
  gemm(mid, EMB_, rw2t, rb2, nullptr, 0, out_reg, 85, B_, 85, EMB_, 0);

  (void)in_sizes; (void)n_in; (void)out_size; (void)ws_size;
}
